// Transformer_13271448945023
// MI455X (gfx1250) — matrix operand order repaired and checked by reading, not compiled, not run on hardware
//
#include <hip/hip_runtime.h>

#define DM  1024
#define NH  16
#define DHD 64
#define BB  8
#define TT  1024
#define BT  (BB*TT)
#define D4  (4*DM)

typedef __attribute__((ext_vector_type(16))) _Float16 v16h;
typedef __attribute__((ext_vector_type(8)))  _Float16 v8h;
typedef __attribute__((ext_vector_type(8)))  float    v8f;
typedef unsigned int u32;
typedef u32 __attribute__((ext_vector_type(4))) u32x4;
typedef int __attribute__((ext_vector_type(8))) i32x8;

// ---- WMMA fragment helpers (layouts per CDNA5 ISA §7.12.2) ----
__device__ __forceinline__ v16h mk16(v8h a, v8h b){
  return __builtin_shufflevector(a, b, 0,1,2,3,4,5,6,7,8,9,10,11,12,13,14,15);
}
// A 16x32 f16: lane=row, lanes0-15: K{0-7,16-23}, lanes16-31: K{8-15,24-31}
__device__ __forceinline__ v16h frag_a(const _Float16* base, int h01){
  v8h lo = *(const v8h*)(base + 8*h01);
  v8h hi = *(const v8h*)(base + 16 + 8*h01);
  return mk16(lo, hi);
}
// B 32x16 f16 (transposed tile rows = N, contiguous K): lanes0-15 K{0-7,16-23}, lanes16-31 K{8-15,24-31}
__device__ __forceinline__ v16h frag_b(const _Float16* base, int h01){
  v8h lo = *(const v8h*)(base + 8*h01);
  v8h hi = *(const v8h*)(base + 16 + 8*h01);
  return mk16(lo, hi);
}
__device__ __forceinline__ v8f wmma16(v16h a, v16h b, v8f c){
  return __builtin_amdgcn_wmma_f32_16x16x32_f16(false, a, false, b, (short)0, c, false, false);
}

// ---- TDM: issue a 2D f16 tile load (rows x cols, row stride in elements) to LDS ----
// D# layout per CDNA5 ISA (group0) / §8.4 (group1).
// 2D tensor -> groups 2/3 disabled (2-operand form). Tracked by TENSORcnt.
__device__ __forceinline__ void tdm_load_tile_f16(u32 lds_off, const _Float16* g,
                                                  u32 rows, u32 cols, u32 stride){
  const unsigned long long ga = (unsigned long long)(uintptr_t)g;
  u32x4 g0;
  g0[0] = 1u;                                              // count=1, user mode
  g0[1] = lds_off;                                         // lds_addr (bytes)
  g0[2] = (u32)ga;                                         // global_addr[31:0]
  g0[3] = (u32)((ga >> 32) & 0x1FFFFFFull) | (2u << 30);   // addr[56:32] | type=2
  i32x8 g1;
  g1[0] = (int)0x00010000u;                                // data_size=1 (2 bytes)
  g1[1] = (int)((cols & 0xFFFFu) << 16);                   // tensor_dim0[15:0]
  g1[2] = (int)((cols >> 16) | ((rows & 0xFFFFu) << 16));  // tdim0 hi | tdim1 lo
  g1[3] = (int)((rows >> 16) | (cols << 16));              // tdim1 hi | tile_dim0
  g1[4] = (int)rows;                                       // tile_dim1 (tile_dim2=0)
  g1[5] = (int)stride;                                     // tensor_dim0_stride[31:0]
  g1[6] = 0;                                               // stride hi | stride1 lo
  g1[7] = 0;
  asm volatile("tensor_load_to_lds %0, %1" :: "s"(g0), "s"(g1) : "memory");
}

// ---- LayerNorm (f32 in -> f16 out), one row (D=1024) per 256-thread block ----
__global__ __launch_bounds__(256) void ln_f16_k(const float* __restrict__ in,
                                                const float* __restrict__ gam,
                                                const float* __restrict__ bet,
                                                _Float16* __restrict__ out){
  const int row = blockIdx.x;
  const float* p = in + (size_t)row * DM;
  float v[4]; float s = 0.f, ss = 0.f;
  #pragma unroll
  for (int i = 0; i < 4; i++){ v[i] = p[threadIdx.x + i*256]; s += v[i]; ss += v[i]*v[i]; }
  #pragma unroll
  for (int off = 16; off > 0; off >>= 1){ s += __shfl_xor(s, off, 32); ss += __shfl_xor(ss, off, 32); }
  __shared__ float sh_s[8], sh_ss[8];
  const int wave = threadIdx.x >> 5, lane = threadIdx.x & 31;
  if (lane == 0){ sh_s[wave] = s; sh_ss[wave] = ss; }
  __syncthreads();
  float S = 0.f, SS = 0.f;
  #pragma unroll
  for (int i = 0; i < 8; i++){ S += sh_s[i]; SS += sh_ss[i]; }
  const float mean = S * (1.f/DM);
  const float var  = SS * (1.f/DM) - mean*mean;
  const float rstd = rsqrtf(var + 1e-5f);
  #pragma unroll
  for (int i = 0; i < 4; i++){
    const int c = threadIdx.x + i*256;
    out[(size_t)row*DM + c] = (_Float16)((v[i]-mean)*rstd*gam[c] + bet[c]);
  }
}

// ---- Weight convert+transpose: W[K][N] f32 -> Wt[N][K] f16 ----
__global__ __launch_bounds__(256) void cvtT_k(const float* __restrict__ W,
                                              _Float16* __restrict__ Wt, int K, int N){
  const size_t i = (size_t)blockIdx.x*256 + threadIdx.x;
  const int n = (int)(i % N); const int kk = (int)(i / N);
  Wt[(size_t)n*K + kk] = (_Float16)W[i];
}

// ---- V transpose per head: v[b][t][h*64+dh] f16 -> vt[b][h][dh][t] f16 ----
__global__ __launch_bounds__(256) void vtrans_k(const _Float16* __restrict__ v,
                                                _Float16* __restrict__ vt){
  const size_t i = (size_t)blockIdx.x*256 + threadIdx.x;
  const int c  = (int)(i % DM);
  const size_t bt = i / DM;
  const int t = (int)(bt % TT); const int b = (int)(bt / TT);
  const int h = c / DHD, dh = c % DHD;
  vt[((size_t)(b*NH + h)*DHD + dh)*TT + t] = v[i];
}

// ---- Tiled WMMA GEMM with TDM double-buffered tile loads ----
// C[M][N] = A[M][K](f16) * Bt[N][K]^T(f16), fused epilogues.
// EPI 0: f16 store.  EPI 1: f32 store + bias + residual.  EPI 2: f16 store of gelu(x+bias).
template<int EPI>
__global__ __launch_bounds__(256) void gemm_wmma_k(
    const _Float16* __restrict__ A, const _Float16* __restrict__ Bt,
    const float* __restrict__ bias, const float* __restrict__ res,
    float* __restrict__ outF, _Float16* __restrict__ outH,
    int M, int N, int K)
{
  __shared__ __align__(16) _Float16 As[2][128*32];
  __shared__ __align__(16) _Float16 Bs[2][128*32];
  const int m0 = blockIdx.y * 128, n0 = blockIdx.x * 128;
  const int tid = threadIdx.x;
  const int lane = tid & 31, wave = tid >> 5;
  const int h01 = lane >> 4, r16 = lane & 15;
  const int wm = wave >> 1, wn = wave & 1;     // 4x2 wave grid, each wave 32Mx64N

  v8f acc[2][4] = {};

  const _Float16* Ab = A  + (size_t)m0 * K;    // tile row origin
  const _Float16* Bb = Bt + (size_t)n0 * K;
  const u32 ldsA[2] = { (u32)(uintptr_t)&As[0][0], (u32)(uintptr_t)&As[1][0] };
  const u32 ldsB[2] = { (u32)(uintptr_t)&Bs[0][0], (u32)(uintptr_t)&Bs[1][0] };
  const int nk = K / 32;

  if (tid < 32){                               // wave 0 drives the TDM
    tdm_load_tile_f16(ldsA[0], Ab, 128, 32, (u32)K);
    tdm_load_tile_f16(ldsB[0], Bb, 128, 32, (u32)K);
  }

  for (int i = 0; i < nk; i++){
    const int cur = i & 1;
    if (tid < 32) __builtin_amdgcn_s_wait_tensorcnt(0);   // buf[cur] landed
    __syncthreads();                                       // publish to all waves
    if (tid < 32 && (i + 1) < nk){                         // prefetch next tiles
      const int nxt = (i + 1) & 1;
      tdm_load_tile_f16(ldsA[nxt], Ab + (size_t)(i+1)*32, 128, 32, (u32)K);
      tdm_load_tile_f16(ldsB[nxt], Bb + (size_t)(i+1)*32, 128, 32, (u32)K);
    }
    v16h af[2], bf[4];
    #pragma unroll
    for (int mt = 0; mt < 2; mt++) af[mt] = frag_a(&As[cur][(wm*32 + mt*16 + r16)*32], h01);
    #pragma unroll
    for (int nt = 0; nt < 4; nt++) bf[nt] = frag_b(&Bs[cur][(wn*64 + nt*16 + r16)*32], h01);
    #pragma unroll
    for (int mt = 0; mt < 2; mt++)
      #pragma unroll
      for (int nt = 0; nt < 4; nt++)
        acc[mt][nt] = wmma16(af[mt], bf[nt], acc[mt][nt]);
  }

  #pragma unroll
  for (int mt = 0; mt < 2; mt++)
    #pragma unroll
    for (int nt = 0; nt < 4; nt++)
      #pragma unroll
      for (int j = 0; j < 8; j++){
        const int gm = m0 + wm*32 + mt*16 + j + 8*h01;
        const int gn = n0 + wn*64 + nt*16 + r16;
        const float vv = acc[mt][nt][j];
        if (EPI == 0){
          outH[(size_t)gm*N + gn] = (_Float16)vv;
        } else if (EPI == 1){
          outF[(size_t)gm*N + gn] = vv + bias[gn] + res[(size_t)gm*N + gn];
        } else {
          const float xg = vv + bias[gn];
          const float t  = tanhf(0.7978845608028654f * (xg + 0.044715f*xg*xg*xg));
          outH[(size_t)gm*N + gn] = (_Float16)(0.5f * xg * (1.0f + t));
        }
      }
}

// ---- Flash attention: block = 8 waves, each wave owns a 16-row q tile (128 q rows/block) ----
__global__ __launch_bounds__(256) void attn_k(
    const _Float16* __restrict__ q, const _Float16* __restrict__ k,
    const _Float16* __restrict__ vt, _Float16* __restrict__ z)
{
  __shared__ __align__(16) _Float16 Ks[32*64];      // [key][dh]
  __shared__ __align__(16) _Float16 Vs[64*32];      // [dh][key]  (from vt)
  __shared__ __align__(16) _Float16 Ps[8][16*32];   // per-wave P tile [row][key]

  const int qblocks = TT/128;
  const int bh   = blockIdx.x / qblocks;
  const int qblk = blockIdx.x % qblocks;
  const int b = bh / NH, h = bh % NH;
  const int tid = threadIdx.x, lane = tid & 31, wave = tid >> 5;
  const int h01 = lane >> 4, r16 = lane & 15;
  const int q0 = qblk*128 + wave*16;
  const size_t bT = (size_t)b * TT;

  v16h qf[2];
  #pragma unroll
  for (int ks = 0; ks < 2; ks++)
    qf[ks] = frag_a(q + (bT + q0 + r16)*DM + h*DHD + ks*32, h01);

  v8f acc[4] = {};
  float runm[8], runl[8];
  #pragma unroll
  for (int j = 0; j < 8; j++){ runm[j] = -1e30f; runl[j] = 0.f; }

  const int nkv = (qblk + 1) * 4;            // 32-key tiles up to block causal edge
  for (int step = 0; step < nkv; step++){
    const int kv0 = step * 32;
    {
      const int r = tid >> 3, sgm = tid & 7;     // K tile: 32 rows x 8 segs
      *(v8h*)&Ks[r*64 + sgm*8] = *(const v8h*)(k + (bT + kv0 + r)*DM + h*DHD + sgm*8);
      const int r2 = tid >> 2, sg2 = tid & 3;    // Vt tile: 64 rows x 4 segs
      *(v8h*)&Vs[r2*32 + sg2*8] = *(const v8h*)(vt + ((size_t)(b*NH + h)*DHD + r2)*TT + kv0 + sg2*8);
    }
    __syncthreads();

    v8f s[2];
    #pragma unroll
    for (int nt = 0; nt < 2; nt++){
      v8f sz = {};
      v16h b0 = frag_b(&Ks[(nt*16 + r16)*64 +  0], h01);
      v16h b1 = frag_b(&Ks[(nt*16 + r16)*64 + 32], h01);
      sz = wmma16(qf[0], b0, sz);
      sz = wmma16(qf[1], b1, sz);
      s[nt] = sz;
    }

    float p0[8], p1[8], tmax[8], rsum[8];
    #pragma unroll
    for (int j = 0; j < 8; j++){
      const int m = q0 + j + 8*h01;
      #pragma unroll
      for (int nt = 0; nt < 2; nt++){
        const int n = kv0 + nt*16 + r16;
        const float sv = s[nt][j] * 0.125f;          // 1/sqrt(64)
        s[nt][j] = (n > m) ? -__builtin_inff() : sv; // causal mask
      }
      tmax[j] = fmaxf(s[0][j], s[1][j]);
    }
    #pragma unroll
    for (int off = 1; off < 16; off <<= 1)
      #pragma unroll
      for (int j = 0; j < 8; j++)
        tmax[j] = fmaxf(tmax[j], __shfl_xor(tmax[j], off, 16));
    #pragma unroll
    for (int j = 0; j < 8; j++){
      const float nm   = fmaxf(runm[j], tmax[j]);
      const float corr = __expf(runm[j] - nm);
      runm[j] = nm; runl[j] *= corr;
      #pragma unroll
      for (int dt = 0; dt < 4; dt++) acc[dt][j] *= corr;
      p0[j] = __expf(s[0][j] - nm);
      p1[j] = __expf(s[1][j] - nm);
      rsum[j] = p0[j] + p1[j];
    }
    #pragma unroll
    for (int off = 1; off < 16; off <<= 1)
      #pragma unroll
      for (int j = 0; j < 8; j++)
        rsum[j] += __shfl_xor(rsum[j], off, 16);
    #pragma unroll
    for (int j = 0; j < 8; j++) runl[j] += rsum[j];

    // C-layout -> A-fragment reshape of P through per-wave LDS
    #pragma unroll
    for (int j = 0; j < 8; j++){
      Ps[wave][(j + 8*h01)*32 + r16]      = (_Float16)p0[j];
      Ps[wave][(j + 8*h01)*32 + 16 + r16] = (_Float16)p1[j];
    }
    asm volatile("s_wait_dscnt 0" ::: "memory");
    v16h pf = frag_a(&Ps[wave][r16*32], h01);

    #pragma unroll
    for (int dt = 0; dt < 4; dt++){
      v16h bv = frag_b(&Vs[(dt*16 + r16)*32], h01);
      acc[dt] = wmma16(pf, bv, acc[dt]);
    }
    __syncthreads();
  }

  #pragma unroll
  for (int j = 0; j < 8; j++){
    const float inv = 1.0f / runl[j];
    const size_t row = bT + q0 + j + 8*h01;
    #pragma unroll
    for (int dt = 0; dt < 4; dt++)
      z[row*DM + h*DHD + dt*16 + r16] = (_Float16)(acc[dt][j] * inv);
  }
}

// ---- Host orchestration ----
extern "C" void kernel_launch(void* const* d_in, const int* in_sizes, int n_in,
                              void* d_out, int out_size, void* d_ws, size_t ws_size,
                              hipStream_t stream) {
  (void)in_sizes; (void)n_in; (void)out_size; (void)ws_size;
  const float* x    = (const float*)d_in[0];
  const float* ln1s = (const float*)d_in[1];
  const float* ln1b = (const float*)d_in[2];
  const float* Wq   = (const float*)d_in[3];
  const float* Wk   = (const float*)d_in[4];
  const float* Wv   = (const float*)d_in[5];
  const float* Wo   = (const float*)d_in[6];
  const float* bo   = (const float*)d_in[7];
  const float* ln2s = (const float*)d_in[8];
  const float* ln2b = (const float*)d_in[9];
  const float* W1   = (const float*)d_in[10];
  const float* b1   = (const float*)d_in[11];
  const float* W2   = (const float*)d_in[12];
  const float* b2   = (const float*)d_in[13];
  float* out = (float*)d_out;

  char* w = (char*)d_ws;
  const size_t MiB = 1ull << 20;
  _Float16* Wqt = (_Float16*)(w + 0*MiB);
  _Float16* Wkt = (_Float16*)(w + 2*MiB);
  _Float16* Wvt = (_Float16*)(w + 4*MiB);
  _Float16* Wot = (_Float16*)(w + 6*MiB);
  _Float16* W1t = (_Float16*)(w + 8*MiB);
  _Float16* W2t = (_Float16*)(w + 16*MiB);
  _Float16* x1h = (_Float16*)(w + 24*MiB);
  _Float16* qh  = (_Float16*)(w + 40*MiB);
  _Float16* kh  = (_Float16*)(w + 56*MiB);
  _Float16* vh  = (_Float16*)(w + 72*MiB);
  _Float16* vtp = (_Float16*)(w + 88*MiB);
  _Float16* zh  = (_Float16*)(w + 104*MiB);
  float*    y   = (float*)   (w + 120*MiB);
  _Float16* y1h = (_Float16*)(w + 152*MiB);
  _Float16* hh  = (_Float16*)(w + 168*MiB);   // total ~232 MiB

  // LN1 + weight conversion
  ln_f16_k<<<BT, 256, 0, stream>>>(x, ln1s, ln1b, x1h);
  cvtT_k<<<(DM*DM)/256, 256, 0, stream>>>(Wq, Wqt, DM, DM);
  cvtT_k<<<(DM*DM)/256, 256, 0, stream>>>(Wk, Wkt, DM, DM);
  cvtT_k<<<(DM*DM)/256, 256, 0, stream>>>(Wv, Wvt, DM, DM);
  cvtT_k<<<(DM*DM)/256, 256, 0, stream>>>(Wo, Wot, DM, DM);
  cvtT_k<<<(DM*D4)/256, 256, 0, stream>>>(W1, W1t, DM, D4);
  cvtT_k<<<(D4*DM)/256, 256, 0, stream>>>(W2, W2t, D4, DM);

  // QKV projections
  dim3 gDD(DM/128, BT/128);
  gemm_wmma_k<0><<<gDD, 256, 0, stream>>>(x1h, Wqt, nullptr, nullptr, nullptr, qh, BT, DM, DM);
  gemm_wmma_k<0><<<gDD, 256, 0, stream>>>(x1h, Wkt, nullptr, nullptr, nullptr, kh, BT, DM, DM);
  gemm_wmma_k<0><<<gDD, 256, 0, stream>>>(x1h, Wvt, nullptr, nullptr, nullptr, vh, BT, DM, DM);
  vtrans_k<<<(size_t)(BT)*DM/256, 256, 0, stream>>>(vh, vtp);

  // Flash attention
  attn_k<<<BB*NH*(TT/128), 256, 0, stream>>>(qh, kh, vtp, zh);

  // Output projection + residual -> y (f32)
  gemm_wmma_k<1><<<gDD, 256, 0, stream>>>(zh, Wot, bo, x, y, nullptr, BT, DM, DM);

  // LN2 -> MLP with fused GELU -> second residual into d_out
  ln_f16_k<<<BT, 256, 0, stream>>>(y, ln2s, ln2b, y1h);
  dim3 gD4(D4/128, BT/128);
  gemm_wmma_k<2><<<gD4, 256, 0, stream>>>(y1h, W1t, b1, nullptr, nullptr, hh, BT, D4, DM);
  gemm_wmma_k<1><<<gDD, 256, 0, stream>>>(hh, W2t, b2, y, out, nullptr, BT, DM, D4);
}
